// Model_15908558865525
// MI455X (gfx1250) — matrix operand order repaired and checked by reading, not compiled, not run on hardware
//
#include <hip/hip_runtime.h>

typedef __attribute__((ext_vector_type(16))) _Float16 v16h;
typedef __attribute__((ext_vector_type(8)))  _Float16 v8h;
typedef __attribute__((ext_vector_type(8)))  float    v8f;

#define B_ 2
#define H_ 16
#define S_ 2048
#define D_ 64
#define DROP_P 0.1f
#define KEEP_SCALE (1.0f / 0.9f)
#define LOG2E 1.44269504088896340736f

#define KC 64                 // keys per chunk
#define WAVES 8
#define QPB (WAVES * 16)      // 128 query rows per block
#define NEG_BIG 1.0e30f

// padded LDS row strides (halfs): 64 data + 16 pad -> 160B rows, keeps 32B
// alignment for v16h operand loads while breaking 128B-stride bank patterns
#define KTS 80
#define VLS 80
#define PLS 80

// v_exp_f32 is a base-2 exponential: use the raw builtin (1 trans op).
__device__ __forceinline__ float fast_exp2(float x) {
  return __builtin_amdgcn_exp2f(x);
}

// ---- DPP16 butterfly reductions across each 16-lane half (pure VALU) ----
// quad_perm xor1 = 0xB1, quad_perm xor2 = 0x4E, row_half_mirror = 0x141,
// row_mirror = 0x140. After the two quad xors all lanes of a quad are equal,
// so the mirror permutations combine quads/octets correctly.
template <int CTRL>
__device__ __forceinline__ float dpp_bcast(float x) {
  int i = __float_as_int(x);
  int r = __builtin_amdgcn_update_dpp(i, i, CTRL, 0xf, 0xf, true);
  return __int_as_float(r);
}
__device__ __forceinline__ float rowmax16(float v) {
  v = fmaxf(v, dpp_bcast<0xB1>(v));
  v = fmaxf(v, dpp_bcast<0x4E>(v));
  v = fmaxf(v, dpp_bcast<0x141>(v));
  v = fmaxf(v, dpp_bcast<0x140>(v));
  return v;
}
__device__ __forceinline__ float rowsum16(float v) {
  v += dpp_bcast<0xB1>(v);
  v += dpp_bcast<0x4E>(v);
  v += dpp_bcast<0x141>(v);
  v += dpp_bcast<0x140>(v);
  return v;
}

__global__ __launch_bounds__(256, 2)
void fa_fwd_dropout(const float* __restrict__ Q, const float* __restrict__ K,
                    const float* __restrict__ V, const float* __restrict__ U,
                    float* __restrict__ O) {
  __shared__ __align__(64) _Float16 Kt[D_ * KTS];          // [key][d]
  __shared__ __align__(64) _Float16 Vl[KC * VLS];          // [d][key] transposed
  __shared__ __align__(64) _Float16 Pl[WAVES * 16 * PLS];  // per-wave 16x64 P

  const int tid  = threadIdx.x;
  const int wave = tid >> 5;
  const int lane = tid & 31;
  const int hi   = lane >> 4;   // lane half
  const int n    = lane & 15;   // C-layout column / A-layout row
  const int c    = hi * 8;      // A-layout K sub-offset per lane half is 2*c

  const int bh     = blockIdx.x;
  const int q_base = blockIdx.y * QPB + wave * 16;

  // ---- load this wave's Q tile into the two A operands (f32 -> f16) ----
  // log2(e) folded into Q so scores are in the log2 domain: softmax is
  // shift-invariant there and v_exp_f32 is base-2 -> one trans op per exp.
  const float* qrow = Q + ((size_t)bh * S_ + (size_t)(q_base + n)) * D_;
  v16h a0, a1;
#pragma unroll
  for (int e = 0; e < 8; ++e) {
    a0[e]     = (_Float16)(qrow[2 * c + e]      * LOG2E);
    a0[e + 8] = (_Float16)(qrow[2 * c + 8 + e]  * LOG2E);
    a1[e]     = (_Float16)(qrow[32 + 2 * c + e] * LOG2E);
    a1[e + 8] = (_Float16)(qrow[40 + 2 * c + e] * LOG2E);
  }

  float m[8], l[8];
  v8f o0 = {}, o1 = {}, o2 = {}, o3 = {};
#pragma unroll
  for (int r = 0; r < 8; ++r) { m[r] = -NEG_BIG; l[r] = 0.f; }

  const size_t kv_base = (size_t)bh * S_ * D_;
  const size_t u_base  = ((size_t)bh * S_ + (size_t)q_base) * (size_t)S_;
  _Float16* pw = &Pl[wave * 16 * PLS];

  for (int k0 = 0; k0 < S_; k0 += KC) {
    __syncthreads();
    // stage K and V (transposed) chunk, f32 -> f16, coalesced global reads
    for (int i = tid; i < KC * D_; i += 256) {
      int key = i >> 6, d = i & 63;
      Kt[key * KTS + d] = (_Float16)K[kv_base + (size_t)(k0 + key) * D_ + d];
      Vl[d * VLS + key] = (_Float16)V[kv_base + (size_t)(k0 + key) * D_ + d];
    }
    if (k0 + KC < S_) {
      __builtin_prefetch(&K[kv_base + (size_t)(k0 + KC) * D_ + (size_t)tid * 16], 0, 1);
      __builtin_prefetch(&V[kv_base + (size_t)(k0 + KC) * D_ + (size_t)tid * 16], 0, 1);
    }
    __syncthreads();

    // ---- S = Q @ K^T : four 16x16 score tiles, K-dim split 64 = 32+32 ----
    v8f s0 = {}, s1 = {}, s2 = {}, s3 = {};
    {
      v16h bl  = *(const v16h*)&Kt[n * KTS + 2 * c];
      v16h bh2 = *(const v16h*)&Kt[n * KTS + 32 + 2 * c];
      s0 = __builtin_amdgcn_wmma_f32_16x16x32_f16(false, a0, false, bl,  (short)0, s0, false, false);
      s0 = __builtin_amdgcn_wmma_f32_16x16x32_f16(false, a1, false, bh2, (short)0, s0, false, false);
    }
    {
      v16h bl  = *(const v16h*)&Kt[(16 + n) * KTS + 2 * c];
      v16h bh2 = *(const v16h*)&Kt[(16 + n) * KTS + 32 + 2 * c];
      s1 = __builtin_amdgcn_wmma_f32_16x16x32_f16(false, a0, false, bl,  (short)0, s1, false, false);
      s1 = __builtin_amdgcn_wmma_f32_16x16x32_f16(false, a1, false, bh2, (short)0, s1, false, false);
    }
    {
      v16h bl  = *(const v16h*)&Kt[(32 + n) * KTS + 2 * c];
      v16h bh2 = *(const v16h*)&Kt[(32 + n) * KTS + 32 + 2 * c];
      s2 = __builtin_amdgcn_wmma_f32_16x16x32_f16(false, a0, false, bl,  (short)0, s2, false, false);
      s2 = __builtin_amdgcn_wmma_f32_16x16x32_f16(false, a1, false, bh2, (short)0, s2, false, false);
    }
    {
      v16h bl  = *(const v16h*)&Kt[(48 + n) * KTS + 2 * c];
      v16h bh2 = *(const v16h*)&Kt[(48 + n) * KTS + 32 + 2 * c];
      s3 = __builtin_amdgcn_wmma_f32_16x16x32_f16(false, a0, false, bl,  (short)0, s3, false, false);
      s3 = __builtin_amdgcn_wmma_f32_16x16x32_f16(false, a1, false, bh2, (short)0, s3, false, false);
    }

    // ---- online softmax (log2 domain) + dropout, C layout row = r+8*hi ----
#pragma unroll
    for (int r = 0; r < 8; ++r) {
      float t  = fmaxf(fmaxf(s0[r], s1[r]), fmaxf(s2[r], s3[r]));
      t = rowmax16(t);
      float mn   = fmaxf(m[r], t);
      float corr = fast_exp2(m[r] - mn);
      float p0 = fast_exp2(s0[r] - mn);
      float p1 = fast_exp2(s1[r] - mn);
      float p2 = fast_exp2(s2[r] - mn);
      float p3 = fast_exp2(s3[r] - mn);
      float rs = rowsum16((p0 + p1) + (p2 + p3));  // denom uses UNdropped p
      l[r] = l[r] * corr + rs;
      m[r] = mn;

      // dropout keep-mask from precomputed uniforms (the 536MB stream)
      const float* urow = U + u_base + (size_t)(r + 8 * hi) * (size_t)S_ + k0;
      p0 *= (urow[n]      >= DROP_P) ? KEEP_SCALE : 0.f;
      p1 *= (urow[16 + n] >= DROP_P) ? KEEP_SCALE : 0.f;
      p2 *= (urow[32 + n] >= DROP_P) ? KEEP_SCALE : 0.f;
      p3 *= (urow[48 + n] >= DROP_P) ? KEEP_SCALE : 0.f;

      int prow = (r + 8 * hi) * PLS;
      pw[prow + n]      = (_Float16)p0;
      pw[prow + 16 + n] = (_Float16)p1;
      pw[prow + 32 + n] = (_Float16)p2;
      pw[prow + 48 + n] = (_Float16)p3;

      o0[r] *= corr; o1[r] *= corr; o2[r] *= corr; o3[r] *= corr;
    }

    // ---- A operands (P, 16x64 f16 -> two 16x32) back from LDS ----
    v16h pa0, pa1;
    {
      const _Float16* pr = pw + n * PLS;   // row M = lane&15
      v8h q0 = *(const v8h*)&pr[2 * c];
      v8h q1 = *(const v8h*)&pr[8 + 2 * c];
      v8h q2 = *(const v8h*)&pr[32 + 2 * c];
      v8h q3 = *(const v8h*)&pr[40 + 2 * c];
#pragma unroll
      for (int e = 0; e < 8; ++e) {
        pa0[e] = q0[e]; pa0[e + 8] = q1[e];
        pa1[e] = q2[e]; pa1[e + 8] = q3[e];
      }
    }

    // ---- O += P @ V : keys split 64 = 32+32, N sub-tiles over D=64 ----
    {
      v16h vb;
      vb = *(const v16h*)&Vl[n * VLS + 2 * c];
      o0 = __builtin_amdgcn_wmma_f32_16x16x32_f16(false, pa0, false, vb, (short)0, o0, false, false);
      vb = *(const v16h*)&Vl[n * VLS + 32 + 2 * c];
      o0 = __builtin_amdgcn_wmma_f32_16x16x32_f16(false, pa1, false, vb, (short)0, o0, false, false);

      vb = *(const v16h*)&Vl[(16 + n) * VLS + 2 * c];
      o1 = __builtin_amdgcn_wmma_f32_16x16x32_f16(false, pa0, false, vb, (short)0, o1, false, false);
      vb = *(const v16h*)&Vl[(16 + n) * VLS + 32 + 2 * c];
      o1 = __builtin_amdgcn_wmma_f32_16x16x32_f16(false, pa1, false, vb, (short)0, o1, false, false);

      vb = *(const v16h*)&Vl[(32 + n) * VLS + 2 * c];
      o2 = __builtin_amdgcn_wmma_f32_16x16x32_f16(false, pa0, false, vb, (short)0, o2, false, false);
      vb = *(const v16h*)&Vl[(32 + n) * VLS + 32 + 2 * c];
      o2 = __builtin_amdgcn_wmma_f32_16x16x32_f16(false, pa1, false, vb, (short)0, o2, false, false);

      vb = *(const v16h*)&Vl[(48 + n) * VLS + 2 * c];
      o3 = __builtin_amdgcn_wmma_f32_16x16x32_f16(false, pa0, false, vb, (short)0, o3, false, false);
      vb = *(const v16h*)&Vl[(48 + n) * VLS + 32 + 2 * c];
      o3 = __builtin_amdgcn_wmma_f32_16x16x32_f16(false, pa1, false, vb, (short)0, o3, false, false);
    }
  }

  // ---- normalize by softmax denominator and store ----
#pragma unroll
  for (int r = 0; r < 8; ++r) {
    float invl = 1.0f / l[r];
    float* orow = O + ((size_t)bh * S_ + (size_t)(q_base + r + 8 * hi)) * D_;
    orow[n]      = o0[r] * invl;
    orow[16 + n] = o1[r] * invl;
    orow[32 + n] = o2[r] * invl;
    orow[48 + n] = o3[r] * invl;
  }
}

extern "C" void kernel_launch(void* const* d_in, const int* in_sizes, int n_in,
                              void* d_out, int out_size, void* d_ws, size_t ws_size,
                              hipStream_t stream) {
  const float* Q = (const float*)d_in[0];
  const float* K = (const float*)d_in[1];
  const float* V = (const float*)d_in[2];
  const float* U = (const float*)d_in[3];
  float* O = (float*)d_out;

  dim3 grid(B_ * H_, S_ / QPB);   // (32, 16)
  fa_fwd_dropout<<<grid, 256, 0, stream>>>(Q, K, V, U, O);
}
